// EagleAttention_46797963657561
// MI455X (gfx1250) — matrix operand order repaired and checked by reading, not compiled, not run on hardware
//
#include <hip/hip_runtime.h>

// ---------------------------------------------------------------------------
// EagleAttention for MI455X (gfx1250, wave32, WMMA, async global->LDS).
// Pipeline: cast->bf16, fused QKV WMMA GEMM, RoPE+reorder, flash attention
// with WMMA (QK^T and P*V), output-projection WMMA GEMM (f32 out).
// ---------------------------------------------------------------------------

typedef __attribute__((ext_vector_type(16))) __bf16 bf16x16;
typedef __attribute__((ext_vector_type(8)))  float  f32x8;

#define B_   2
#define S_   2048
#define HID_ 2048
#define H_   16
#define KV_  8
#define D_   128
#define M_   (B_ * S_)          // 4096 rows of (b,s)
#define K0_  (2 * HID_)         // 4096 input dim
#define NCAT_ 4096              // q(2048) | k(1024) | v(1024)
#define SCALE_ 0.08838834764831845f   // 1/sqrt(128)

// ---------------- async global -> LDS staging (CDNA5) -----------------------

#ifndef __has_builtin
#define __has_builtin(x) 0
#endif

#if __has_builtin(__builtin_amdgcn_global_load_async_to_lds_b128)
#define USE_ASYNC_BUILTIN 1
#else
#define USE_ASYNC_BUILTIN 0
#endif

#if USE_ASYNC_BUILTIN
typedef __attribute__((__vector_size__(4 * sizeof(int)))) int v4i_t;
typedef __attribute__((address_space(1))) v4i_t* g_v4i_ptr;
typedef __attribute__((address_space(3))) v4i_t* l_v4i_ptr;
#endif

__device__ __forceinline__ void async_b128(void* ldsp, const void* gp) {
#if USE_ASYNC_BUILTIN
  __builtin_amdgcn_global_load_async_to_lds_b128(
      (g_v4i_ptr)(gp), (l_v4i_ptr)(ldsp), 0, 0);
#else
  // ISA mnemonic fallback: vdst = LDS byte address, vaddr = 64-bit global.
  unsigned l =
      (unsigned)(unsigned long long)(__attribute__((address_space(3))) char*)
          ldsp;
  asm volatile("global_load_async_to_lds_b128 %0, %1, off" ::"v"(l), "v"(gp)
               : "memory");
#endif
}

__device__ __forceinline__ void async_wait() {
  asm volatile("s_wait_asynccnt 0x0" ::: "memory");
}

// ---------------- WMMA helpers (CDNA5 layouts, wave32) ----------------------

__device__ __forceinline__ f32x8 wmma_bf16(bf16x16 a, bf16x16 b, f32x8 c) {
  return __builtin_amdgcn_wmma_f32_16x16x32_bf16(
      /*neg_a=*/false, a, /*neg_b=*/false, b,
      /*c_mod=*/(short)0, c, /*reuse_a=*/false, /*reuse_b=*/false);
}

// A-matrix 16x32 bf16 fragment from row-major storage (lda elems per row).
// ISA layout: lanes 0-15 row=lane, elems = K[0..7],K[16..23];
//             lanes 16-31 row=lane-16, elems = K[8..15],K[24..31].
__device__ __forceinline__ bf16x16 load_a_frag(const __bf16* base, int lda) {
  const int lane = threadIdx.x & 31;
  const int row  = lane & 15;
  const int kb   = (lane < 16) ? 0 : 8;
  union { bf16x16 v; uint4 u[2]; } r;
  r.u[0] = *(const uint4*)(base + row * lda + kb);
  r.u[1] = *(const uint4*)(base + row * lda + kb + 16);
  return r.v;
}

// B-matrix 32x16 bf16 fragment from K-contiguous transposed storage Bt[n][k]
// (ldb elems per n-row). Lanes 0-15: col=lane, K 0..7,16..23; lanes 16-31: K 8..15,24..31.
__device__ __forceinline__ bf16x16 load_b_frag(const __bf16* baseT, int ldb) {
  const int lane = threadIdx.x & 31;
  const int col  = lane & 15;
  const int kb   = (lane < 16) ? 0 : 8;
  union { bf16x16 v; uint4 u[2]; } r;
  r.u[0] = *(const uint4*)(baseT + col * ldb + kb);
  r.u[1] = *(const uint4*)(baseT + col * ldb + kb + 16);
  return r.v;
}

// ---------------- Stage 1: casts / weight concat ----------------------------

__global__ void __launch_bounds__(256)
cast_f32_to_bf16_x8(const float* __restrict__ src, __bf16* __restrict__ dst,
                    int n8) {
  int i = blockIdx.x * blockDim.x + threadIdx.x;
  if (i >= n8) return;
  const float4* s = (const float4*)src + (long)i * 2;
  float4 a = s[0], b = s[1];
  union { uint4 u; __bf16 h[8]; } o;
  o.h[0] = (__bf16)a.x; o.h[1] = (__bf16)a.y;
  o.h[2] = (__bf16)a.z; o.h[3] = (__bf16)a.w;
  o.h[4] = (__bf16)b.x; o.h[5] = (__bf16)b.y;
  o.h[6] = (__bf16)b.z; o.h[7] = (__bf16)b.w;
  ((uint4*)dst)[i] = o.u;
}

__global__ void __launch_bounds__(256)
build_wcat(const float* __restrict__ Wq, const float* __restrict__ Wk,
           const float* __restrict__ Wv, __bf16* __restrict__ Wcat) {
  int i = blockIdx.x * blockDim.x + threadIdx.x;   // over 4096*4096 elems
  int c = i & (NCAT_ - 1);
  int r = i >> 12;
  float v;
  if (c < 2048)       v = Wq[(long)r * 2048 + c];
  else if (c < 3072)  v = Wk[(long)r * 1024 + (c - 2048)];
  else                v = Wv[(long)r * 1024 + (c - 3072)];
  Wcat[(long)r * NCAT_ + c] = (__bf16)v;
}

// ---------------- Stage 2/5: generic bf16 WMMA GEMM -------------------------
// C[M,N] = A[M,K] * B[K,N], 128x128 block tile, BK=32, 8 waves per block.

template <bool OUT_F32>
__global__ void __launch_bounds__(256)
gemm_bf16_wmma(const __bf16* __restrict__ A, const __bf16* __restrict__ Bm,
               void* __restrict__ Cout, int M, int N, int K) {
  __shared__ __align__(16) __bf16 As[128 * 32];
  __shared__ __align__(16) __bf16 Bt[128 * 32];   // transposed: Bt[n][k]

  const int tid  = threadIdx.x;
  const int lane = tid & 31;
  const int w    = tid >> 5;
  const int wm   = w & 3;        // 4 waves along M
  const int wn   = w >> 2;       // 2 waves along N
  const int m0   = blockIdx.y * 128;
  const int n0   = blockIdx.x * 128;

  f32x8 acc[2][4];
#pragma unroll
  for (int i = 0; i < 2; ++i)
#pragma unroll
    for (int j = 0; j < 4; ++j) acc[i][j] = (f32x8){0,0,0,0,0,0,0,0};

  const int arow  = tid >> 1;
  const int ahalf = (tid & 1) * 16;
  const int bk    = tid & 31;
  const int bn    = (tid >> 5) * 16;

  for (int k0 = 0; k0 < K; k0 += 32) {
    // stage A tile (128x32) via async global->LDS DMA
    {
      const __bf16* s = A + (long)(m0 + arow) * K + k0 + ahalf;
      __bf16* d = As + arow * 32 + ahalf;
      async_b128(d, s);
      async_b128(d + 8, s + 8);
    }
    // stage B tile (32x128) transposed into Bt[n][k] (register transpose)
    {
      union { uint4 u; __bf16 h[8]; } t0, t1;
      const uint4* s = (const uint4*)(Bm + (long)(k0 + bk) * N + n0 + bn);
      t0.u = s[0];
      t1.u = s[1];
      if (k0 + 32 < K)
        __builtin_prefetch(Bm + (long)(k0 + 32 + bk) * N + n0 + bn, 0, 3);
#pragma unroll
      for (int e = 0; e < 8; ++e) {
        Bt[(bn + e) * 32 + bk]     = t0.h[e];
        Bt[(bn + 8 + e) * 32 + bk] = t1.h[e];
      }
    }
    async_wait();
    __syncthreads();

    bf16x16 af0 = load_a_frag(As + (wm * 32) * 32, 32);
    bf16x16 af1 = load_a_frag(As + (wm * 32 + 16) * 32, 32);
    bf16x16 bfr[4];
#pragma unroll
    for (int j = 0; j < 4; ++j)
      bfr[j] = load_b_frag(Bt + (wn * 64 + j * 16) * 32, 32);
#pragma unroll
    for (int j = 0; j < 4; ++j) {
      acc[0][j] = wmma_bf16(af0, bfr[j], acc[0][j]);
      acc[1][j] = wmma_bf16(af1, bfr[j], acc[1][j]);
    }
    __syncthreads();
  }

  const int col0  = lane & 15;
  const int rbase = (lane < 16) ? 0 : 8;
#pragma unroll
  for (int i = 0; i < 2; ++i) {
#pragma unroll
    for (int j = 0; j < 4; ++j) {
      const int cg = n0 + wn * 64 + j * 16 + col0;
#pragma unroll
      for (int r = 0; r < 8; ++r) {
        const long rg = m0 + wm * 32 + i * 16 + rbase + r;
        if (OUT_F32)
          ((float*)Cout)[rg * N + cg] = acc[i][j][r];
        else
          ((__bf16*)Cout)[rg * N + cg] = (__bf16)acc[i][j][r];
      }
    }
  }
}

// ---------------- Stage 3: RoPE + GQA reorder -------------------------------
// qkv_raw: [B*S, 4096] bf16 (q|k|v). Outputs: Qr[B,H,S,D], Kr/Vr[B,KV,S,D].

#define NQP_ (B_ * S_ * H_ * 64)    // 4194304 q rotation pairs
#define NKP_ (B_ * S_ * KV_ * 64)   // 2097152 k rotation pairs
#define NV_  (B_ * S_ * KV_ * D_)   // 4194304 v elements

__global__ void __launch_bounds__(256)
rope_reorder(const __bf16* __restrict__ qkv, __bf16* __restrict__ Qr,
             __bf16* __restrict__ Kr, __bf16* __restrict__ Vr) {
  int i = blockIdx.x * blockDim.x + threadIdx.x;
  const float kln = 9.210340371976184f / 64.f;   // ln(10000)/64
  if (i < NQP_) {
    int d = i & 63; int t = i >> 6;
    int h = t & 15; t >>= 4;
    int s = t & 2047; int b = t >> 11;
    const __bf16* row = qkv + (long)(b * S_ + s) * NCAT_ + h * D_;
    float x0 = (float)row[d], x1 = (float)row[d + 64];
    float f = __expf(-(float)d * kln);
    float sn, cs; __sincosf((float)s * f, &sn, &cs);
    __bf16* out = Qr + ((long)(b * H_ + h) * S_ + s) * D_;
    out[d]      = (__bf16)(x0 * cs - x1 * sn);
    out[d + 64] = (__bf16)(x1 * cs + x0 * sn);
  } else if (i < NQP_ + NKP_) {
    int j = i - NQP_;
    int d = j & 63; int t = j >> 6;
    int kh = t & 7; t >>= 3;
    int s = t & 2047; int b = t >> 11;
    const __bf16* row = qkv + (long)(b * S_ + s) * NCAT_ + 2048 + kh * D_;
    float x0 = (float)row[d], x1 = (float)row[d + 64];
    float f = __expf(-(float)d * kln);
    float sn, cs; __sincosf((float)s * f, &sn, &cs);
    __bf16* out = Kr + ((long)(b * KV_ + kh) * S_ + s) * D_;
    out[d]      = (__bf16)(x0 * cs - x1 * sn);
    out[d + 64] = (__bf16)(x1 * cs + x0 * sn);
  } else if (i < NQP_ + NKP_ + NV_) {
    int j = i - NQP_ - NKP_;
    int d = j & 127; int t = j >> 7;
    int kh = t & 7; t >>= 3;
    int s = t & 2047; int b = t >> 11;
    Vr[((long)(b * KV_ + kh) * S_ + s) * D_ + d] =
        qkv[(long)(b * S_ + s) * NCAT_ + 3072 + kh * D_ + d];
  }
}

// ---------------- Stage 4: causal flash attention (WMMA) --------------------
// Block: 128 query rows for one (b,h). 8 waves x 16 rows. Key tiles of 64.

__global__ void __launch_bounds__(256)
flash_attn_wmma(const __bf16* __restrict__ Qr, const __bf16* __restrict__ Kr,
                const __bf16* __restrict__ Vr, __bf16* __restrict__ Aout) {
  __shared__ __align__(16) __bf16 Qs[128 * 128];  // [qrow][d]     32 KB
  __shared__ __align__(16) __bf16 Ks[64 * 128];   // [key][d]      16 KB
  __shared__ __align__(16) __bf16 Vt[128 * 64];   // [d][key]      16 KB
  __shared__ __align__(16) __bf16 Ps[8 * 16 * 64];// per-wave P    16 KB

  const int tid  = threadIdx.x;
  const int lane = tid & 31;
  const int w    = tid >> 5;
  const int bh   = blockIdx.y;
  const int b    = bh / H_;
  const int h    = bh % H_;
  const int kvh  = h >> 1;                 // n_rep = 2
  const int qt   = blockIdx.x;
  const int q0   = qt * 128;

  const __bf16* Qg = Qr + ((long)(b * H_ + h) * S_ + q0) * D_;
  const __bf16* Kg = Kr + ((long)(b * KV_ + kvh) * S_) * D_;
  const __bf16* Vg = Vr + ((long)(b * KV_ + kvh) * S_) * D_;

  // load Q tile via async DMA: 2 threads per row, 8x b128 each
  {
    const int row = tid >> 1, half = (tid & 1) * 64;
    const __bf16* s = Qg + (long)row * D_ + half;
    __bf16* d = Qs + row * D_ + half;
#pragma unroll
    for (int i = 0; i < 8; ++i) async_b128(d + i * 8, s + i * 8);
  }

  f32x8 acc_o[8];
#pragma unroll
  for (int t = 0; t < 8; ++t) acc_o[t] = (f32x8){0,0,0,0,0,0,0,0};
  float mr[8], lr[8];
#pragma unroll
  for (int r = 0; r < 8; ++r) { mr[r] = -1e30f; lr[r] = 0.f; }

  const int col0  = lane & 15;
  const int rbase = (lane < 16) ? 0 : 8;
  __bf16* Ps_w = Ps + w * 16 * 64;

  const int nkb = 2 * qt + 2;              // causal: keys <= q0+127
  for (int kb = 0; kb < nkb; ++kb) {
    __syncthreads();                       // prior compute done
    // stage K tile [64 keys][128 d] via async DMA
    {
      const int row = tid >> 2, q4 = (tid & 3) * 32;
      const __bf16* s = Kg + (long)(kb * 64 + row) * D_ + q4;
      __bf16* d = Ks + row * D_ + q4;
#pragma unroll
      for (int i = 0; i < 4; ++i) async_b128(d + i * 8, s + i * 8);
    }
    // stage V tile transposed -> Vt[d][key] (register transpose)
    {
      const int key = tid >> 2, db = (tid & 3) * 32;
      union { uint4 u; __bf16 hh[8]; } t4[4];
      const uint4* s = (const uint4*)(Vg + (long)(kb * 64 + key) * D_ + db);
      t4[0].u = s[0]; t4[1].u = s[1]; t4[2].u = s[2]; t4[3].u = s[3];
#pragma unroll
      for (int q = 0; q < 4; ++q)
#pragma unroll
        for (int e = 0; e < 8; ++e)
          Vt[(db + q * 8 + e) * 64 + key] = t4[q].hh[e];
    }
    async_wait();
    __syncthreads();

    // S = Q K^T : wave rows [w*16, w*16+16), 4 key tiles of 16
    f32x8 sacc[4];
#pragma unroll
    for (int j = 0; j < 4; ++j) sacc[j] = (f32x8){0,0,0,0,0,0,0,0};
#pragma unroll
    for (int kc = 0; kc < 4; ++kc) {
      bf16x16 af = load_a_frag(Qs + (w * 16) * D_ + kc * 32, D_);
      bf16x16 bfr[4];
#pragma unroll
      for (int j = 0; j < 4; ++j)
        bfr[j] = load_b_frag(Ks + (j * 16) * D_ + kc * 32, D_);
#pragma unroll
      for (int j = 0; j < 4; ++j) sacc[j] = wmma_bf16(af, bfr[j], sacc[j]);
    }

    // scale + causal mask
#pragma unroll
    for (int j = 0; j < 4; ++j) {
      const int key = kb * 64 + j * 16 + col0;
#pragma unroll
      for (int r = 0; r < 8; ++r) {
        const int qrow = q0 + w * 16 + rbase + r;
        float sc = sacc[j][r] * SCALE_;
        sacc[j][r] = (key <= qrow) ? sc : -1e30f;
      }
    }

    // online softmax per row (16-lane group reductions)
#pragma unroll
    for (int r = 0; r < 8; ++r) {
      float v = fmaxf(fmaxf(sacc[0][r], sacc[1][r]),
                      fmaxf(sacc[2][r], sacc[3][r]));
      v = fmaxf(v, __shfl_xor(v, 1, 32));
      v = fmaxf(v, __shfl_xor(v, 2, 32));
      v = fmaxf(v, __shfl_xor(v, 4, 32));
      v = fmaxf(v, __shfl_xor(v, 8, 32));
      const float mnew  = fmaxf(mr[r], v);
      const float alpha = __expf(mr[r] - mnew);
      mr[r] = mnew;
      lr[r] *= alpha;
#pragma unroll
      for (int t = 0; t < 8; ++t) acc_o[t][r] = acc_o[t][r] * alpha;
      float rs = 0.f;
#pragma unroll
      for (int j = 0; j < 4; ++j) {
        float p = __expf(sacc[j][r] - mnew);
        sacc[j][r] = p;
        rs += p;
      }
      rs += __shfl_xor(rs, 1, 32);
      rs += __shfl_xor(rs, 2, 32);
      rs += __shfl_xor(rs, 4, 32);
      rs += __shfl_xor(rs, 8, 32);
      lr[r] += rs;
    }

    // transpose P through per-wave LDS into A-fragment form
#pragma unroll
    for (int j = 0; j < 4; ++j)
#pragma unroll
      for (int r = 0; r < 8; ++r)
        Ps_w[(rbase + r) * 64 + j * 16 + col0] = (__bf16)sacc[j][r];

    // O += P V  (8 d-tiles x 2 key chunks), B-frags preloaded in groups of 4
#pragma unroll
    for (int kc = 0; kc < 2; ++kc) {
      bf16x16 af = load_a_frag(Ps_w + kc * 32, 64);
#pragma unroll
      for (int th = 0; th < 2; ++th) {
        bf16x16 bfr[4];
#pragma unroll
        for (int j = 0; j < 4; ++j)
          bfr[j] =
              load_b_frag(Vt + ((th * 4 + j) * 16) * 64 + kc * 32, 64);
#pragma unroll
        for (int j = 0; j < 4; ++j)
          acc_o[th * 4 + j] = wmma_bf16(af, bfr[j], acc_o[th * 4 + j]);
      }
    }
  }

  // normalize + store to attn_out[B*S, H*D] (bf16)
#pragma unroll
  for (int t = 0; t < 8; ++t) {
    const int cg = h * D_ + t * 16 + col0;
#pragma unroll
    for (int r = 0; r < 8; ++r) {
      const long rg = (long)b * S_ + q0 + w * 16 + rbase + r;
      Aout[rg * (H_ * D_) + cg] = (__bf16)(acc_o[t][r] / lr[r]);
    }
  }
}

// ---------------------------------------------------------------------------

extern "C" void kernel_launch(void* const* d_in, const int* in_sizes, int n_in,
                              void* d_out, int out_size, void* d_ws,
                              size_t ws_size, hipStream_t stream) {
  (void)in_sizes; (void)n_in; (void)out_size; (void)ws_size;
  const float* hs = (const float*)d_in[0];
  // d_in[1] attention_mask (causal, applied analytically)
  // d_in[2] position_ids   (arange, applied analytically)
  const float* Wq = (const float*)d_in[3];
  const float* Wk = (const float*)d_in[4];
  const float* Wv = (const float*)d_in[5];
  const float* Wo = (const float*)d_in[6];
  float* out = (float*)d_out;

  char* ws = (char*)d_ws;
  __bf16* hs_bf  = (__bf16*)(ws);                        // 32 MB
  __bf16* Wcat   = (__bf16*)(ws + (32ll << 20));         // 32 MB
  __bf16* qkvraw = (__bf16*)(ws + (64ll << 20));         // 32 MB
  __bf16* Wo_bf  = (__bf16*)(ws + (96ll << 20));         //  8 MB
  // region reuse after consumers finish:
  __bf16* Qr   = (__bf16*)(ws);                          // 16 MB (over hs_bf)
  __bf16* Kr   = (__bf16*)(ws + (16ll << 20));           //  8 MB
  __bf16* Vr   = (__bf16*)(ws + (24ll << 20));           //  8 MB
  __bf16* Aout = (__bf16*)(ws + (64ll << 20));           // 16 MB (over qkvraw)

  // 1) casts + weight concat
  {
    int n8 = (M_ * K0_) / 8;
    cast_f32_to_bf16_x8<<<n8 / 256, 256, 0, stream>>>(hs, hs_bf, n8);
  }
  {
    int n8 = (HID_ * HID_) / 8;
    cast_f32_to_bf16_x8<<<n8 / 256, 256, 0, stream>>>(Wo, Wo_bf, n8);
  }
  build_wcat<<<(K0_ * NCAT_) / 256, 256, 0, stream>>>(Wq, Wk, Wv, Wcat);

  // 2) fused QKV projection: [4096,4096] x [4096,4096] -> bf16
  gemm_bf16_wmma<false><<<dim3(NCAT_ / 128, M_ / 128), 256, 0, stream>>>(
      hs_bf, Wcat, (void*)qkvraw, M_, NCAT_, K0_);

  // 3) RoPE + reorder into [B,H,S,D] / [B,KV,S,D]
  {
    int total = NQP_ + NKP_ + NV_;
    rope_reorder<<<(total + 255) / 256, 256, 0, stream>>>(qkvraw, Qr, Kr, Vr);
  }

  // 4) causal flash attention with WMMA
  flash_attn_wmma<<<dim3(S_ / 128, B_ * H_), 256, 0, stream>>>(Qr, Kr, Vr,
                                                               Aout);

  // 5) output projection: [4096,2048] x [2048,2048] -> f32 d_out
  gemm_bf16_wmma<true><<<dim3(HID_ / 128, M_ / 128), 256, 0, stream>>>(
      Aout, Wo_bf, (void*)out, M_, HID_, H_ * D_);
}
